// MultiHeadAttention_19731079758119
// MI455X (gfx1250) — matrix operand order repaired and checked by reading, not compiled, not run on hardware
//
#include <hip/hip_runtime.h>
#include <hip/hip_bf16.h>

// MHA (flux-style RoPE) for B=2,S=2048,D=1024,H=16,HD=64 on gfx1250.
// bf16 WMMA (v_wmma_f32_16x16x32_bf16) for all four GEMMs + both attention
// einsums, fp32 accumulate/softmax. Tile staging via Tensor Data Mover
// (tensor_load_to_lds, TENSORcnt) with double buffering when available.

#define BB   2
#define SS   2048
#define DD   1024
#define HH   16
#define HDIM 64
#define MM   (BB * SS)   // 4096 tokens

typedef __attribute__((ext_vector_type(16))) __bf16 v16bf;
typedef __attribute__((ext_vector_type(8)))  float  v8f;

union FragB { unsigned int u[8]; v16bf v; };

__device__ __forceinline__ unsigned short f2bf(float f) {
  unsigned int u = __float_as_uint(f);
  unsigned int r = (u + 0x7fffu + ((u >> 16) & 1u)) >> 16;   // RNE
  return (unsigned short)r;
}
__device__ __forceinline__ unsigned int pack2bf(float a, float b) {
  return (unsigned int)f2bf(a) | ((unsigned int)f2bf(b) << 16);
}

// ---------------- Tensor Data Mover support ----------------
#if defined(__has_builtin)
#if __has_builtin(__builtin_amdgcn_tensor_load_to_lds) && \
    __has_builtin(__builtin_amdgcn_s_wait_tensorcnt)
#define HAVE_TDM 1
#endif
#endif
#ifndef HAVE_TDM
#define HAVE_TDM 0
#endif

#if HAVE_TDM
typedef __attribute__((ext_vector_type(4))) unsigned int v4u_t;
typedef __attribute__((ext_vector_type(8))) int          v8i_t;
typedef __attribute__((ext_vector_type(4))) int          v4i_t;

__device__ __forceinline__ unsigned lds_off32(const void* p) {
  // flat shared addr: aperture in [63:32], LDS byte offset in [31:0]
  return (unsigned)(unsigned long long)(size_t)p;
}

// 2D tile load global->LDS via TDM. Sizes in bf16 elements (data_size=2B).
// pad_int_code: pad after 2^(code+1) dwords; pad_amt_code: (code+1) dwords.
__device__ __forceinline__ void tdm_load_2d(unsigned lds_byte_off, const void* gptr,
                                            unsigned tile_w, unsigned tile_h,
                                            unsigned row_stride_elems,
                                            unsigned pad_int_code, unsigned pad_amt_code,
                                            unsigned pad_enable) {
  unsigned long long ga = (unsigned long long)(size_t)gptr;
  v4u_t g0;
  g0[0] = 1u;                                             // count=1 (valid user D#)
  g0[1] = lds_byte_off;                                   // lds_addr
  g0[2] = (unsigned)ga;                                   // global_addr[31:0]
  g0[3] = (unsigned)((ga >> 32) & 0x01ffffffu) | 0x80000000u; // addr[56:32]|type=2
  v8i_t g1;
  g1[0] = (int)((1u << 16) | (pad_enable << 20) |         // data_size=1 (2B)
                (pad_int_code << 22) | (pad_amt_code << 25));
  g1[1] = (int)((tile_w & 0xffffu) << 16);                // tensor_dim0 lo16
  g1[2] = (int)((tile_w >> 16) | ((tile_h & 0xffffu) << 16)); // dim0 hi | dim1 lo
  g1[3] = (int)((tile_h >> 16) | ((tile_w & 0xffffu) << 16)); // dim1 hi | tile_dim0
  g1[4] = (int)(tile_h & 0xffffu);                        // tile_dim1 (tile_dim2=0)
  g1[5] = (int)row_stride_elems;                          // tensor_dim0_stride[31:0]
  g1[6] = 0;
  g1[7] = 0;
  v4i_t z4 = {0, 0, 0, 0};
#if __clang_major__ >= 23
  v8i_t z8 = {0, 0, 0, 0, 0, 0, 0, 0};
  __builtin_amdgcn_tensor_load_to_lds(g0, g1, z4, z4, z8, 0);
#else
  __builtin_amdgcn_tensor_load_to_lds(g0, g1, z4, z4, 0);
#endif
}
#define TDM_WAIT() __builtin_amdgcn_s_wait_tensorcnt(0)
#else
#define TDM_WAIT() ((void)0)
#endif

// ---------------- fp32 -> packed bf16 ----------------
__global__ __launch_bounds__(256)
void cvt_f32_bf16(const float* __restrict__ src, unsigned int* __restrict__ dst, int n2) {
  int i = blockIdx.x * blockDim.x + threadIdx.x;
  if (i < n2) dst[i] = pack2bf(src[2 * i], src[2 * i + 1]);
}

// ============ shared GEMM core: 128(M) x 64(N) x 32(K) tiles ============
// A[m,k], W[n,k] row-major bf16 (dword pairs); C tile 16x16 per wave per nt.
// LDS rows padded to 20 dwords (TDM pad: 16 dwords data + 4 pad).

#define GEMM_COMPUTE(LA, LB)                                                   \
  {                                                                            \
    FragB fa;                                                                  \
    _Pragma("unroll") for (int v = 0; v < 8; ++v) {                            \
      int kd = v + 4 * ((v >> 2) + hi); /* A 16x32 bf16 per-lane layout */     \
      fa.u[v] = (LA)[wave * 16 + lo][kd];                                      \
    }                                                                          \
    _Pragma("unroll") for (int nt = 0; nt < 4; ++nt) {                         \
      FragB fb;                                                                \
      _Pragma("unroll") for (int v = 0; v < 8; ++v)                            \
          fb.u[v] = (LB)[nt * 16 + lo][v + 4 * ((v >> 2) + hi)]; /* B 32x16 layout */ \
      acc[nt] = __builtin_amdgcn_wmma_f32_16x16x32_bf16(                       \
          false, fa.v, false, fb.v, (short)0, acc[nt], false, false);          \
    }                                                                          \
  }

// ---------------- fused QKV GEMM + bias + RoPE ----------------
__global__ __launch_bounds__(256)
void qkv_rope_kernel(const unsigned int* __restrict__ xb,   // [4096][512] dwords
                     const unsigned int* __restrict__ wqb,
                     const unsigned int* __restrict__ wkb,
                     const unsigned int* __restrict__ wvb,
                     const float* __restrict__ bq,
                     const float* __restrict__ bk,
                     const float* __restrict__ bv,
                     const float* __restrict__ freqs,       // [S][32][2][2] fp32
                     unsigned short* __restrict__ qbuf,     // [B,H,S,HD] bf16
                     unsigned short* __restrict__ kbuf,
                     unsigned short* __restrict__ vbuf) {
  const int which = blockIdx.z;
  const unsigned int* wb   = (which == 0) ? wqb : (which == 1) ? wkb : wvb;
  const float*        bias = (which == 0) ? bq  : (which == 1) ? bk  : bv;
  unsigned short*     outb = (which == 0) ? qbuf : (which == 1) ? kbuf : vbuf;

  __shared__ unsigned int ldsA[2][128][20];   // double-buffered A tiles
  __shared__ unsigned int ldsB[2][64][20];

  const int tid  = threadIdx.x;
  const int wave = tid >> 5, lane = tid & 31;
  const int lo = lane & 15, hi = lane >> 4;
  const int mBase = blockIdx.y * 128;
  const int nBase = blockIdx.x * 64;

  v8f acc[4] = {};

#if HAVE_TDM
  // prologue: tile 0 into buffer 0 (one TDM per producer wave)
  if (wave == 0)
    tdm_load_2d(lds_off32(&ldsA[0][0][0]), xb + (size_t)mBase * 512,
                32, 128, DD, 3, 3, 1);
  if (wave == 1)
    tdm_load_2d(lds_off32(&ldsB[0][0][0]), wb + (size_t)nBase * 512,
                32, 64, DD, 3, 3, 1);
  TDM_WAIT();
  __syncthreads();
#endif

  for (int kb = 0; kb < DD / 32; ++kb) {
    const int cur = kb & 1;
#if HAVE_TDM
    if (kb + 1 < DD / 32) {   // async DMA for next tile overlaps this compute
      const int nxt = cur ^ 1, kd1 = (kb + 1) * 16;
      if (wave == 0)
        tdm_load_2d(lds_off32(&ldsA[nxt][0][0]), xb + (size_t)mBase * 512 + kd1,
                    32, 128, DD, 3, 3, 1);
      if (wave == 1)
        tdm_load_2d(lds_off32(&ldsB[nxt][0][0]), wb + (size_t)nBase * 512 + kd1,
                    32, 64, DD, 3, 3, 1);
    }
#else
    {   // manual staging fallback
      const int kd0 = kb * 16;
      int r = tid >> 1, h8 = (tid & 1) * 8;
      const uint4* src = (const uint4*)(xb + (size_t)(mBase + r) * 512 + kd0 + h8);
      uint4 a0 = src[0], a1 = src[1];
      uint4* dstp = (uint4*)&ldsA[cur][r][h8];
      dstp[0] = a0; dstp[1] = a1;
      int r2 = tid >> 2, q4 = (tid & 3) * 4;
      uint4 b0 = *(const uint4*)(wb + (size_t)(nBase + r2) * 512 + kd0 + q4);
      *(uint4*)&ldsB[cur][r2][q4] = b0;
      __syncthreads();
    }
#endif
    GEMM_COMPUTE(ldsA[cur], ldsB[cur]);
    TDM_WAIT();
    __syncthreads();
  }

  const bool dorope = (which != 2);
#pragma unroll
  for (int nt = 0; nt < 4; ++nt) {
    int col = nBase + nt * 16 + lo;
    int h = col >> 6, d = col & 63;
    float bval = bias[col];
#pragma unroll
    for (int j = 0; j < 8; ++j) {
      int m = mBase + wave * 16 + j + 8 * hi;    // C layout: row = j + 8*hi
      int b = m >> 11, s = m & (SS - 1);
      float val = acc[nt][j] + bval;
      float res = val;
      if (dorope) {
        float part = __shfl_xor(val, 1, 32);     // pair partner = adjacent lane
        int p = d >> 1, r = d & 1;
        const float* f = freqs + ((((size_t)s * 32 + p) * 2 + r) * 2);
        float t0 = r ? part : val;
        float t1 = r ? val  : part;
        res = f[0] * t0 + f[1] * t1;
      }
      size_t oidx = ((((size_t)b * HH + h) * SS) + s) * HDIM + d;
      outb[oidx] = f2bf(res);
    }
  }
}

// ---------------- flash attention ----------------
// One block = one (b,h) x 128 query rows; 8 waves x 16 rows; 32-key steps.
// K tiles via TDM (padded rows), V tiles manually transposed; double-buffered.
__global__ __launch_bounds__(256)
void flash_attn_kernel(const unsigned short* __restrict__ qg,   // [B*H][S][64] bf16
                       const unsigned short* __restrict__ kg,
                       const unsigned short* __restrict__ vg,
                       unsigned short* __restrict__ obuf) {     // [B,S,D] bf16
  __shared__ unsigned int ldsK[2][32][36];     // 32 keys x 32 dwords (+4 pad)
  __shared__ unsigned int ldsVt[2][64][17];    // V^T: 64 d-rows x 16 dwords
  __shared__ unsigned int ldsP[8][16][17];     // per-wave P staging (16x32 bf16)

  const int bh = blockIdx.x;
  const int b = bh >> 4, h = bh & 15;
  const int qBase = blockIdx.y * 128;
  const int tid = threadIdx.x;
  const int wave = tid >> 5, lane = tid & 31;
  const int lo = lane & 15, hi = lane >> 4;

  const unsigned int* qrow = (const unsigned int*)qg + (size_t)bh * SS * 32;
  const unsigned int* krow = (const unsigned int*)kg + (size_t)bh * SS * 32;
  const unsigned int* vrow = (const unsigned int*)vg + (size_t)bh * SS * 32;

  // Q fragments (A layout), HD=64 -> two k=32 chunks, loaded once.
  FragB qf[2];
  const int qm = qBase + wave * 16 + lo;
#pragma unroll
  for (int c = 0; c < 2; ++c)
#pragma unroll
    for (int v = 0; v < 8; ++v) {
      int kd = v + 4 * ((v >> 2) + hi) + 16 * c;
      qf[c].u[v] = qrow[(size_t)qm * 32 + kd];
    }

  v8f oacc[4] = {};
  float mrow[8], lrow[8];
#pragma unroll
  for (int j = 0; j < 8; ++j) { mrow[j] = -3.0e38f; lrow[j] = 0.f; }

  const int NB = SS / 32;

#define STAGE_V(KB, BUF)                                                       \
  {                                                                            \
    int key = tid >> 3, grp = tid & 7;                                         \
    uint4 v4 = *(const uint4*)(vrow + (size_t)((KB) * 32 + key) * 32 + grp * 4);\
    unsigned int uu[4] = {v4.x, v4.y, v4.z, v4.w};                             \
    int d0 = grp * 8;                                                          \
    _Pragma("unroll") for (int i = 0; i < 4; ++i) {                            \
      ((unsigned short*)&ldsVt[BUF][d0 + 2 * i][0])[key] =                     \
          (unsigned short)(uu[i] & 0xffffu);                                   \
      ((unsigned short*)&ldsVt[BUF][d0 + 2 * i + 1][0])[key] =                 \
          (unsigned short)(uu[i] >> 16);                                       \
    }                                                                          \
  }

#if HAVE_TDM
  STAGE_V(0, 0);
  if (wave == 0)
    tdm_load_2d(lds_off32(&ldsK[0][0][0]), krow, 64, 32, 64, 4, 3, 1);
  TDM_WAIT();
  __syncthreads();
#endif

  for (int kb = 0; kb < NB; ++kb) {
    const int cur = kb & 1;
#if HAVE_TDM
    if (kb + 1 < NB) {
      STAGE_V(kb + 1, cur ^ 1);
      if (wave == 0)
        tdm_load_2d(lds_off32(&ldsK[cur ^ 1][0][0]),
                    krow + (size_t)(kb + 1) * 32 * 32, 64, 32, 64, 4, 3, 1);
    }
#else
    {   // manual K staging fallback
      int r = tid >> 3, oct = (tid & 7) * 4;
      uint4 k4 = *(const uint4*)(krow + (size_t)(kb * 32 + r) * 32 + oct);
      ldsK[cur][r][oct + 0] = k4.x; ldsK[cur][r][oct + 1] = k4.y;
      ldsK[cur][r][oct + 2] = k4.z; ldsK[cur][r][oct + 3] = k4.w;
    }
    STAGE_V(kb, cur);
    if (kb + 1 < NB)
      __builtin_prefetch(krow + (size_t)(kb + 1) * 32 * 32 + (tid & 7) * 64, 0, 1);
    __syncthreads();
#endif

    // scores: S = Q(16x64) @ K_tile^T(64x32); two 16-key subtiles, k=32 chains
    v8f sc[2] = {};
#pragma unroll
    for (int nt = 0; nt < 2; ++nt)
#pragma unroll
      for (int c = 0; c < 2; ++c) {
        FragB fb;
#pragma unroll
        for (int v = 0; v < 8; ++v)
          fb.u[v] = ldsK[cur][nt * 16 + lo][v + 4 * ((v >> 2) + hi) + 16 * c];
        sc[nt] = __builtin_amdgcn_wmma_f32_16x16x32_bf16(
            false, qf[c].v, false, fb.v, (short)0, sc[nt], false, false);
      }

    // online softmax; row = j + 8*hi, stats replicated across 16-lane halves
#pragma unroll
    for (int j = 0; j < 8; ++j) {
      float s0 = sc[0][j] * 0.125f, s1 = sc[1][j] * 0.125f;  // 1/sqrt(64)
      float mj = fmaxf(s0, s1);
      mj = fmaxf(mj, __shfl_xor(mj, 1, 32));
      mj = fmaxf(mj, __shfl_xor(mj, 2, 32));
      mj = fmaxf(mj, __shfl_xor(mj, 4, 32));
      mj = fmaxf(mj, __shfl_xor(mj, 8, 32));
      float mnew  = fmaxf(mrow[j], mj);
      float alpha = __expf(mrow[j] - mnew);
      mrow[j] = mnew;
      float p0 = __expf(s0 - mnew), p1 = __expf(s1 - mnew);
      float rs = p0 + p1;
      rs += __shfl_xor(rs, 1, 32);
      rs += __shfl_xor(rs, 2, 32);
      rs += __shfl_xor(rs, 4, 32);
      rs += __shfl_xor(rs, 8, 32);
      lrow[j] = lrow[j] * alpha + rs;
#pragma unroll
      for (int dt = 0; dt < 4; ++dt) oacc[dt][j] *= alpha;
      unsigned short* pr = (unsigned short*)&ldsP[wave][j + 8 * hi][0];
      pr[lo]      = f2bf(p0);      // C layout -> row-major 16x32 in LDS
      pr[16 + lo] = f2bf(p1);
    }
    asm volatile("s_wait_dscnt 0" ::: "memory");  // P stores visible to reload

    // O += P(16x32) @ V(32x64)
    FragB pf;
#pragma unroll
    for (int v = 0; v < 8; ++v) {
      int kd = v + 4 * ((v >> 2) + hi);
      pf.u[v] = ldsP[wave][lo][kd];
    }
#pragma unroll
    for (int dt = 0; dt < 4; ++dt) {
      FragB fb;
#pragma unroll
      for (int v = 0; v < 8; ++v) fb.u[v] = ldsVt[cur][dt * 16 + lo][v + 4 * ((v >> 2) + hi)];
      oacc[dt] = __builtin_amdgcn_wmma_f32_16x16x32_bf16(
          false, pf.v, false, fb.v, (short)0, oacc[dt], false, false);
    }
    TDM_WAIT();
    __syncthreads();
  }
#undef STAGE_V

  // normalize + write bf16 [B,S,D] for the output projection
#pragma unroll
  for (int j = 0; j < 8; ++j) {
    float inv = 1.0f / lrow[j];
    int srow = qBase + wave * 16 + j + 8 * hi;
    size_t row = (size_t)b * SS + srow;
#pragma unroll
    for (int dt = 0; dt < 4; ++dt) {
      int col = h * 64 + dt * 16 + lo;
      obuf[row * DD + col] = f2bf(oacc[dt][j] * inv);
    }
  }
}

// ---------------- output projection: fp32 out = O @ Wo^T + bo ----------------
__global__ __launch_bounds__(256)
void out_proj_kernel(const unsigned int* __restrict__ ab,   // [4096][512] dwords
                     const unsigned int* __restrict__ wob,
                     const float* __restrict__ bo,
                     float* __restrict__ out) {
  __shared__ unsigned int ldsA[2][128][20];
  __shared__ unsigned int ldsB[2][64][20];

  const int tid  = threadIdx.x;
  const int wave = tid >> 5, lane = tid & 31;
  const int lo = lane & 15, hi = lane >> 4;
  const int mBase = blockIdx.y * 128;
  const int nBase = blockIdx.x * 64;

  v8f acc[4] = {};

#if HAVE_TDM
  if (wave == 0)
    tdm_load_2d(lds_off32(&ldsA[0][0][0]), ab + (size_t)mBase * 512,
                32, 128, DD, 3, 3, 1);
  if (wave == 1)
    tdm_load_2d(lds_off32(&ldsB[0][0][0]), wob + (size_t)nBase * 512,
                32, 64, DD, 3, 3, 1);
  TDM_WAIT();
  __syncthreads();
#endif

  for (int kb = 0; kb < DD / 32; ++kb) {
    const int cur = kb & 1;
#if HAVE_TDM
    if (kb + 1 < DD / 32) {
      const int nxt = cur ^ 1, kd1 = (kb + 1) * 16;
      if (wave == 0)
        tdm_load_2d(lds_off32(&ldsA[nxt][0][0]), ab + (size_t)mBase * 512 + kd1,
                    32, 128, DD, 3, 3, 1);
      if (wave == 1)
        tdm_load_2d(lds_off32(&ldsB[nxt][0][0]), wob + (size_t)nBase * 512 + kd1,
                    32, 64, DD, 3, 3, 1);
    }
#else
    {
      const int kd0 = kb * 16;
      int r = tid >> 1, h8 = (tid & 1) * 8;
      const uint4* src = (const uint4*)(ab + (size_t)(mBase + r) * 512 + kd0 + h8);
      uint4 a0 = src[0], a1 = src[1];
      uint4* dstp = (uint4*)&ldsA[cur][r][h8];
      dstp[0] = a0; dstp[1] = a1;
      int r2 = tid >> 2, q4 = (tid & 3) * 4;
      uint4 b0 = *(const uint4*)(wob + (size_t)(nBase + r2) * 512 + kd0 + q4);
      *(uint4*)&ldsB[cur][r2][q4] = b0;
      __syncthreads();
    }
#endif
    GEMM_COMPUTE(ldsA[cur], ldsB[cur]);
    TDM_WAIT();
    __syncthreads();
  }

#pragma unroll
  for (int nt = 0; nt < 4; ++nt) {
    int col = nBase + nt * 16 + lo;
    float bval = bo[col];
#pragma unroll
    for (int j = 0; j < 8; ++j) {
      int m = mBase + wave * 16 + j + 8 * hi;
      out[(size_t)m * DD + col] = acc[nt][j] + bval;
    }
  }
}

extern "C" void kernel_launch(void* const* d_in, const int* in_sizes, int n_in,
                              void* d_out, int out_size, void* d_ws, size_t ws_size,
                              hipStream_t stream) {
  (void)in_sizes; (void)n_in; (void)out_size; (void)ws_size;
  const float* x  = (const float*)d_in[0];
  const float* fr = (const float*)d_in[1];
  const float* Wq = (const float*)d_in[2];
  const float* bq = (const float*)d_in[3];
  const float* Wk = (const float*)d_in[4];
  const float* bk = (const float*)d_in[5];
  const float* Wv = (const float*)d_in[6];
  const float* bv = (const float*)d_in[7];
  const float* Wo = (const float*)d_in[8];
  const float* bo = (const float*)d_in[9];
  float* out = (float*)d_out;

  char* ws = (char*)d_ws;
  unsigned int*   xb   = (unsigned int*)(ws);                          //  8 MiB
  unsigned int*   wqb  = (unsigned int*)(ws + (8ull  << 20));          //  2 MiB
  unsigned int*   wkb  = (unsigned int*)(ws + (10ull << 20));
  unsigned int*   wvb  = (unsigned int*)(ws + (12ull << 20));
  unsigned int*   wob  = (unsigned int*)(ws + (14ull << 20));
  unsigned short* qbuf = (unsigned short*)(ws + (16ull << 20));        //  8 MiB
  unsigned short* kbuf = (unsigned short*)(ws + (24ull << 20));
  unsigned short* vbuf = (unsigned short*)(ws + (32ull << 20));
  unsigned short* obuf = (unsigned short*)(ws + (40ull << 20));        //  8 MiB

  int n2 = (MM * DD) / 2;
  cvt_f32_bf16<<<(n2 + 255) / 256, 256, 0, stream>>>(x, xb, n2);
  n2 = (DD * DD) / 2;
  cvt_f32_bf16<<<(n2 + 255) / 256, 256, 0, stream>>>(Wq, wqb, n2);
  cvt_f32_bf16<<<(n2 + 255) / 256, 256, 0, stream>>>(Wk, wkb, n2);
  cvt_f32_bf16<<<(n2 + 255) / 256, 256, 0, stream>>>(Wv, wvb, n2);
  cvt_f32_bf16<<<(n2 + 255) / 256, 256, 0, stream>>>(Wo, wob, n2);

  dim3 g1(DD / 64, MM / 128, 3);
  qkv_rope_kernel<<<g1, 256, 0, stream>>>(xb, wqb, wkb, wvb, bq, bk, bv, fr,
                                          qbuf, kbuf, vbuf);

  dim3 g2(BB * HH, SS / 128, 1);
  flash_attn_kernel<<<g2, 256, 0, stream>>>(qbuf, kbuf, vbuf, obuf);

  dim3 g3(DD / 64, MM / 128, 1);
  out_proj_kernel<<<g3, 256, 0, stream>>>((const unsigned int*)obuf, wob, bo, out);
}
